// PixelEachSubstitutor_86577950752804
// MI455X (gfx1250) — matrix operand order repaired and checked by reading, not compiled, not run on hardware
//
#include <hip/hip_runtime.h>
#include <hip/hip_bf16.h>

// ===========================================================================
// PixelEachSubstitutor, fully fused: one wave32 per pixel (43,200 blocks).
// All dense matmuls -> v_wmma_f32_16x16x32_f16. head_dim==1 attention and
// rank-1 FFNs -> VALU. Cross-attention over a single memory token reduces to
// adding a precomputed constant vector (softmax over one key == 1).
// Memory traffic is trivial (input 1.7MB fits L2); design is occupancy-bound.
// ===========================================================================

typedef __attribute__((ext_vector_type(16))) _Float16 v16h;
typedef __attribute__((ext_vector_type(8)))  float    v8f;

// ---- flat input order: jax pytree flatten (alphabetical dict keys) --------
enum {
  I_C_W1 = 0, I_C_W2, I_C_CA_WK, I_C_CA_WO, I_C_CA_WQ, I_C_CA_WV,
  I_C_LN1, I_C_LN2, I_C_LN3, I_C_SA_WK, I_C_SA_WO, I_C_SA_WQ, I_C_SA_WV,
  I_L_W1, I_L_W2, I_L_CA_WK, I_L_CA_WO, I_L_CA_WQ, I_L_CA_WV,
  I_L_LN1, I_L_LN2, I_L_LN3, I_L_SA_WK, I_L_SA_WO, I_L_SA_WQ, I_L_SA_WV,
  I_DEC_W, I_E_W1, I_E_W2, I_E_LN1, I_E_LN2,
  I_E_SA_WK, I_E_SA_WO, I_E_SA_WQ, I_E_SA_WV,
  I_R_CW, I_R_LW, I_SAMP_LW, I_SAMP_W, I_X, N_IN
};

// ---- workspace layout: f16 packed weights then f32 vectors ---------------
// f16 element offsets
#define H_EWQ   0       // enc Wq  32x16 (11x11 zero-padded, K-pad to 32)
#define H_EWK   512
#define H_EWV   1024
#define H_EWO   1536
#define H_SW    2048    // samp_W  64x32 (49x32 padded, samp_Lw folded in)
#define H_LWQ   4096    // attnL.sa 32x32
#define H_LWK   5120
#define H_LWV   6144
#define H_LWO   7168
#define H_CWQ   8192    // attnC.sa 32x16 (10x10 padded)
#define H_CWK   8704
#define H_CWV   9216
#define H_CWO   9728
#define H_TOTAL 10240
// f32 element offsets (base = d_ws + 2*H_TOTAL bytes)
#define F_ELN1  0
#define F_ELN2  16
#define F_EW1   32
#define F_EW2   48
#define F_LLN1  64
#define F_LLN2  96
#define F_LLN3  128
#define F_LW1   160
#define F_LW2   192
#define F_CAL   224     // r_Lw @ WvL_ca @ WoL_ca  (cross-attn constant, d=32)
#define F_CLN1  256
#define F_CLN2  272
#define F_CLN3  288
#define F_CW1   304
#define F_CW2   320
#define F_CAC   336     // r_Cw @ WvC_ca @ WoC_ca  (cross-attn constant, d=10)
#define F_DECW  352
#define F_TOTAL 384

// ---------------------------------------------------------------------------
// WMMA helpers (wave32). Layouts per CDNA5 ISA §7.12.2:
//  A 16x32 f16 : lane<16 -> row m=lane, K 0-7 & 16-23 ; lane>=16 -> K 8-15 & 24-31
//  B 32x16 f16 : lane&15 -> col n   ; lane<16 -> K 0-7 & 16-23 ; lane>=16 -> K 8-15 & 24-31
//  C/D 16x16 f32: lane&15 -> col n  ; vgpr r -> row r + 8*(lane>>4)
// ---------------------------------------------------------------------------
__device__ __forceinline__ v8f wmma32(v16h a, v16h b, v8f c) {
  return __builtin_amdgcn_wmma_f32_16x16x32_f16(false, a, false, b,
                                                (short)0, c, false, false);
}

__device__ __forceinline__ v16h frag_a(const _Float16* A, int lda, int m0,
                                       int k0, int lane) {
  const int m  = m0 + (lane & 15);
  const int kb = (lane >> 4) << 3;
  v16h a;
#pragma unroll
  for (int i = 0; i < 16; ++i)
    a[i] = A[m * lda + k0 + kb + (i & 7) + ((i >> 3) << 4)];
  return a;
}

__device__ __forceinline__ v16h frag_b(const _Float16* B, int ldb, int k0,
                                       int n0, int lane) {
  const int n  = n0 + (lane & 15);
  const int kb = (lane >> 4) << 3;
  v16h b;
#pragma unroll
  for (int i = 0; i < 16; ++i)
    b[i] = B[(k0 + kb + (i & 7) + ((i >> 3) << 4)) * ldb + n];
  return b;
}

__device__ __forceinline__ void frag_st(float* D, int ldd, int m0, int n0,
                                        int lane, v8f d) {
  const int n  = n0 + (lane & 15);
  const int mb = (lane >> 4) << 3;
#pragma unroll
  for (int r = 0; r < 8; ++r)
    D[(m0 + mb + r) * ldd + n] = d[r];
}

// ===========================================================================
// Kernel 0: pack all parameters into padded f16 tiles + f32 vectors, and
// precompute the two cross-attention constant vectors.
// ===========================================================================
struct PtrPack { const float* p[39]; };

__global__ void pes_prep(PtrPack P, _Float16* __restrict__ wsH,
                         float* __restrict__ wsF) {
  const int tid = threadIdx.x;
  for (int i = tid; i < H_TOTAL; i += blockDim.x) wsH[i] = (_Float16)0.f;
  for (int i = tid; i < F_TOTAL; i += blockDim.x) wsF[i] = 0.f;
  __syncthreads();

  // encoder attention weights 11x11 -> 32x16 (row=K in, col=N out)
  for (int i = tid; i < 121; i += blockDim.x) {
    const int k = i / 11, n = i % 11;
    wsH[H_EWQ + k * 16 + n] = (_Float16)P.p[I_E_SA_WQ][i];
    wsH[H_EWK + k * 16 + n] = (_Float16)P.p[I_E_SA_WK][i];
    wsH[H_EWV + k * 16 + n] = (_Float16)P.p[I_E_SA_WV][i];
    wsH[H_EWO + k * 16 + n] = (_Float16)P.p[I_E_SA_WO][i];
  }
  // sampler 49x32 -> 64x32, fold samp_Lw into rows
  for (int i = tid; i < 49 * 32; i += blockDim.x)
    wsH[H_SW + i] = (_Float16)(P.p[I_SAMP_LW][i / 32] * P.p[I_SAMP_W][i]);
  // attnL self-attn 32x32
  for (int i = tid; i < 1024; i += blockDim.x) {
    wsH[H_LWQ + i] = (_Float16)P.p[I_L_SA_WQ][i];
    wsH[H_LWK + i] = (_Float16)P.p[I_L_SA_WK][i];
    wsH[H_LWV + i] = (_Float16)P.p[I_L_SA_WV][i];
    wsH[H_LWO + i] = (_Float16)P.p[I_L_SA_WO][i];
  }
  // attnC self-attn 10x10 -> 32x16
  for (int i = tid; i < 100; i += blockDim.x) {
    const int k = i / 10, n = i % 10;
    wsH[H_CWQ + k * 16 + n] = (_Float16)P.p[I_C_SA_WQ][i];
    wsH[H_CWK + k * 16 + n] = (_Float16)P.p[I_C_SA_WK][i];
    wsH[H_CWV + k * 16 + n] = (_Float16)P.p[I_C_SA_WV][i];
    wsH[H_CWO + k * 16 + n] = (_Float16)P.p[I_C_SA_WO][i];
  }
  // vectors
  for (int i = tid; i < 11; i += blockDim.x) {
    wsF[F_ELN1 + i] = P.p[I_E_LN1][i];
    wsF[F_ELN2 + i] = P.p[I_E_LN2][i];
    wsF[F_EW1  + i] = P.p[I_E_W1][i];
    wsF[F_EW2  + i] = P.p[I_E_W2][i];
  }
  for (int i = tid; i < 32; i += blockDim.x) {
    wsF[F_LLN1 + i] = P.p[I_L_LN1][i];
    wsF[F_LLN2 + i] = P.p[I_L_LN2][i];
    wsF[F_LLN3 + i] = P.p[I_L_LN3][i];
    wsF[F_LW1  + i] = P.p[I_L_W1][i];
    wsF[F_LW2  + i] = P.p[I_L_W2][i];
    wsF[F_DECW + i] = P.p[I_DEC_W][i];
  }
  for (int i = tid; i < 10; i += blockDim.x) {
    wsF[F_CLN1 + i] = P.p[I_C_LN1][i];
    wsF[F_CLN2 + i] = P.p[I_C_LN2][i];
    wsF[F_CLN3 + i] = P.p[I_C_LN3][i];
    wsF[F_CW1  + i] = P.p[I_C_W1][i];
    wsF[F_CW2  + i] = P.p[I_C_W2][i];
  }
  // cross-attn constants: softmax over a single key == 1 -> (mem@Wv)@Wo
  __shared__ float tL[32], tC[16];
  if (tid < 32) {
    float s = 0.f;
    for (int i = 0; i < 32; ++i) s += P.p[I_R_LW][i] * P.p[I_L_CA_WV][i * 32 + tid];
    tL[tid] = s;
  }
  if (tid >= 32 && tid < 42) {
    const int t = tid - 32; float s = 0.f;
    for (int i = 0; i < 10; ++i) s += P.p[I_R_CW][i] * P.p[I_C_CA_WV][i * 10 + t];
    tC[t] = s;
  }
  __syncthreads();
  if (tid < 32) {
    float s = 0.f;
    for (int j = 0; j < 32; ++j) s += tL[j] * P.p[I_L_CA_WO][j * 32 + tid];
    wsF[F_CAL + tid] = s;
  }
  if (tid >= 32 && tid < 42) {
    const int t = tid - 32; float s = 0.f;
    for (int j = 0; j < 10; ++j) s += tC[j] * P.p[I_C_CA_WO][j * 10 + t];
    wsF[F_CAC + t] = s;
  }
}

// ===========================================================================
// Kernel 1: fused per-pixel pipeline, one wave32 per block.
// ===========================================================================
__global__ __launch_bounds__(32) void pes_fused(
    const _Float16* __restrict__ wsH, const float* __restrict__ wsF,
    const float* __restrict__ xin, float* __restrict__ out) {

  __shared__ _Float16 Xb[64 * 32];   // tokens [64][32], cols 11..31 zero
  __shared__ _Float16 Qb[64 * 32];   // Q / T / Zh+Oh / Uh+Oh scratch
  __shared__ _Float16 Ktb[32 * 64];  // K^T [32][64] / sampler A^T
  __shared__ _Float16 Vb[64 * 16];   // V [64][16]
  __shared__ float    Fb[64 * 64];   // f32 scratch (scores, then tiles)

  const int lane = threadIdx.x;
  const int b  = blockIdx.x;
  const int w0 = b % 30, h0 = (b / 30) % 30, nn = b / 900;
  const float* xg = xin + (size_t)nn * 9000;   // [10][30][30]
  __builtin_prefetch(xg, 0, 0);                // global_prefetch_b8

  for (int i = lane; i < 64 * 32; i += 32) { Xb[i] = (_Float16)0.f; Qb[i] = (_Float16)0.f; }
  for (int i = lane; i < 32 * 64; i += 32) Ktb[i] = (_Float16)0.f;
  for (int i = lane; i < 64 * 16; i += 32) Vb[i]  = (_Float16)0.f;
  __syncthreads();

  // ---- build 49 tokens of d=11 (10 channels + pad-indicator) --------------
  for (int idx = lane; idx < 49 * 11; idx += 32) {
    const int t = idx / 11, c = idx % 11;
    const int hh = h0 + t / 7 - 3, ww = w0 + t % 7 - 3;
    const bool in = (hh >= 0) & (hh < 30) & (ww >= 0) & (ww < 30);
    const float v = (c < 10) ? (in ? xg[(c * 30 + hh) * 30 + ww] : 0.f)
                             : (in ? 0.f : 1.f);
    Xb[t * 32 + c] = (_Float16)v;
  }
  __syncthreads();

  // ---- encoder: QKV projections (Q pre-scaled by 1/sqrt(11)) --------------
  const _Float16 *EWQ = wsH + H_EWQ, *EWK = wsH + H_EWK,
                 *EWV = wsH + H_EWV, *EWO = wsH + H_EWO;
  const float qs = rsqrtf(11.f);
#pragma unroll
  for (int mt = 0; mt < 4; ++mt) {
    v16h a = frag_a(Xb, 32, mt * 16, 0, lane);
    v8f q = {}; q = wmma32(a, frag_b(EWQ, 16, 0, 0, lane), q);
    v8f k = {}; k = wmma32(a, frag_b(EWK, 16, 0, 0, lane), k);
    v8f v = {}; v = wmma32(a, frag_b(EWV, 16, 0, 0, lane), v);
    const int nc = lane & 15, mb = (lane >> 4) << 3;
#pragma unroll
    for (int r = 0; r < 8; ++r) {
      const int m = mt * 16 + mb + r;
      Qb[m * 32 + nc]  = (_Float16)(q[r] * qs);
      Ktb[nc * 64 + m] = (_Float16)k[r];     // store K transposed
      Vb[m * 16 + nc]  = (_Float16)v[r];
    }
  }
  __syncthreads();

  // ---- scores S = Q K^T (64x64 f32) ---------------------------------------
#pragma unroll
  for (int mt = 0; mt < 4; ++mt)
#pragma unroll
    for (int nt = 0; nt < 4; ++nt) {
      v8f s = {};
      s = wmma32(frag_a(Qb, 32, mt * 16, 0, lane),
                 frag_b(Ktb, 64, 0, nt * 16, lane), s);
      frag_st(Fb, 64, mt * 16, nt * 16, lane, s);
    }
  __syncthreads();

  // ---- softmax rows (mask cols >= 49), rewrite in-place as f16 P ----------
  _Float16* P = (_Float16*)Fb;   // f16 row r overwrites f32 row r/2 <= r: safe
  for (int t = 0; t < 64; ++t) {
    float v0 = Fb[t * 64 + lane];
    float v1 = (lane + 32 < 49) ? Fb[t * 64 + 32 + lane] : -1e30f;
    float m = fmaxf(v0, v1);
#pragma unroll
    for (int o = 16; o > 0; o >>= 1) m = fmaxf(m, __shfl_xor(m, o, 32));
    float e0 = __expf(v0 - m), e1 = __expf(v1 - m);
    float s = e0 + e1;
#pragma unroll
    for (int o = 16; o > 0; o >>= 1) s += __shfl_xor(s, o, 32);
    const float inv = 1.f / s;
    P[t * 64 + lane]      = (_Float16)(e0 * inv);
    P[t * 64 + 32 + lane] = (_Float16)(e1 * inv);
  }
  __syncthreads();

  // ---- T = P V -> Qb cols 0..15 -------------------------------------------
#pragma unroll
  for (int mt = 0; mt < 4; ++mt) {
    v8f t = {};
    t = wmma32(frag_a(P, 64, mt * 16,  0, lane), frag_b(Vb, 16,  0, 0, lane), t);
    t = wmma32(frag_a(P, 64, mt * 16, 32, lane), frag_b(Vb, 16, 32, 0, lane), t);
    const int nc = lane & 15, mb = (lane >> 4) << 3;
#pragma unroll
    for (int r = 0; r < 8; ++r)
      Qb[(mt * 16 + mb + r) * 32 + nc] = (_Float16)t[r];
  }
  __syncthreads();

  // ---- O = T Wo -> Fb[64][16] ---------------------------------------------
#pragma unroll
  for (int mt = 0; mt < 4; ++mt) {
    v8f o = {};
    o = wmma32(frag_a(Qb, 32, mt * 16, 0, lane), frag_b(EWO, 16, 0, 0, lane), o);
    frag_st(Fb, 16, mt * 16, 0, lane, o);
  }
  __syncthreads();

  // ---- residual + LN1, then FFN (11->1->11) + LN2 -> Xb -------------------
  const float *ELN1 = wsF + F_ELN1, *ELN2 = wsF + F_ELN2,
              *EW1  = wsF + F_EW1,  *EW2  = wsF + F_EW2;
  for (int t = lane; t < 49; t += 32) {
    float mean = 0.f;
    for (int c = 0; c < 11; ++c) mean += (float)Xb[t * 32 + c] + Fb[t * 16 + c];
    mean *= (1.f / 11.f);
    float var = 0.f;
    for (int c = 0; c < 11; ++c) {
      const float d = (float)Xb[t * 32 + c] + Fb[t * 16 + c] - mean;
      var += d * d;
    }
    const float inv = rsqrtf(var * (1.f / 11.f) + 1e-5f);
    for (int c = 0; c < 11; ++c) {
      const float d = (float)Xb[t * 32 + c] + Fb[t * 16 + c] - mean;
      Xb[t * 32 + c] = (_Float16)(d * inv * ELN1[c]);
    }
  }
  __syncthreads();
  for (int t = lane; t < 49; t += 32) {
    float hs = 0.f;
    for (int c = 0; c < 11; ++c) hs += (float)Xb[t * 32 + c] * EW1[c];
    hs = fmaxf(hs, 0.f);
    float mean = 0.f;
    for (int c = 0; c < 11; ++c) mean += (float)Xb[t * 32 + c] + hs * EW2[c];
    mean *= (1.f / 11.f);
    float var = 0.f;
    for (int c = 0; c < 11; ++c) {
      const float d = (float)Xb[t * 32 + c] + hs * EW2[c] - mean;
      var += d * d;
    }
    const float inv = rsqrtf(var * (1.f / 11.f) + 1e-5f);
    for (int c = 0; c < 11; ++c) {
      const float d = (float)Xb[t * 32 + c] + hs * EW2[c] - mean;
      Xb[t * 32 + c] = (_Float16)(d * inv * ELN2[c]);
    }
  }
  __syncthreads();

  // ---- sampler: Z[10][32] = E^T (10x49) @ sampW (49x32) -------------------
  for (int i = lane; i < 16 * 64; i += 32) Ktb[i] = (_Float16)0.f;
  __syncthreads();
  for (int idx = lane; idx < 10 * 49; idx += 32) {
    const int c = idx / 49, t = idx % 49;
    Ktb[c * 64 + t] = Xb[t * 32 + c];
  }
  __syncthreads();
  const _Float16* SW = wsH + H_SW;
#pragma unroll
  for (int nt = 0; nt < 2; ++nt) {
    v8f z = {};
    z = wmma32(frag_a(Ktb, 64, 0,  0, lane), frag_b(SW, 32,  0, nt * 16, lane), z);
    z = wmma32(frag_a(Ktb, 64, 0, 32, lane), frag_b(SW, 32, 32, nt * 16, lane), z);
    frag_st(Fb, 32, 0, nt * 16, lane, z);    // ZL = Fb[16][32]
  }
  __syncthreads();

  // ======================= reasoner attnL (d=32, seq=10) ===================
  for (int i = lane; i < 16 * 32; i += 32) Qb[i] = (_Float16)Fb[i];  // Zh
  __syncthreads();
  float *QL = Fb + 512, *KL = Fb + 1024, *VL = Fb + 1536, *OL = Fb + 2048;
  const _Float16 *LWQ = wsH + H_LWQ, *LWK = wsH + H_LWK,
                 *LWV = wsH + H_LWV, *LWO = wsH + H_LWO;
#pragma unroll
  for (int nt = 0; nt < 2; ++nt) {
    v16h a = frag_a(Qb, 32, 0, 0, lane);
    v8f q = {}; q = wmma32(a, frag_b(LWQ, 32, 0, nt * 16, lane), q);
    v8f k = {}; k = wmma32(a, frag_b(LWK, 32, 0, nt * 16, lane), k);
    v8f v = {}; v = wmma32(a, frag_b(LWV, 32, 0, nt * 16, lane), v);
    frag_st(QL, 32, 0, nt * 16, lane, q);
    frag_st(KL, 32, 0, nt * 16, lane, k);
    frag_st(VL, 32, 0, nt * 16, lane, v);
  }
  __syncthreads();
  // 32 heads of dim 1: head = lane, scalar q*k attention over 10 keys
  for (int q = 0; q < 10; ++q) {
    const float qv = QL[q * 32 + lane];
    float m = -1e30f;
    for (int k = 0; k < 10; ++k) m = fmaxf(m, qv * KL[k * 32 + lane]);
    float s = 0.f, o = 0.f;
    for (int k = 0; k < 10; ++k) {
      const float e = __expf(qv * KL[k * 32 + lane] - m);
      s += e; o += e * VL[k * 32 + lane];
    }
    OL[q * 32 + lane] = o / s;
  }
#pragma unroll
  for (int q = 10; q < 16; ++q) OL[q * 32 + lane] = 0.f;
  __syncthreads();
  for (int i = lane; i < 16 * 32; i += 32) Qb[512 + i] = (_Float16)OL[i];  // Oh
  __syncthreads();
  float* OW = Fb + 2048;
#pragma unroll
  for (int nt = 0; nt < 2; ++nt) {
    v8f o = {};
    o = wmma32(frag_a(Qb, 32, 16, 0, lane), frag_b(LWO, 32, 0, nt * 16, lane), o);
    frag_st(OW, 32, 0, nt * 16, lane, o);
  }
  __syncthreads();
  const float *LLN1 = wsF + F_LLN1, *LLN2 = wsF + F_LLN2, *LLN3 = wsF + F_LLN3,
              *LW1  = wsF + F_LW1,  *LW2  = wsF + F_LW2,  *CAL  = wsF + F_CAL;
  if (lane < 10) {
    const int t = lane;
    // self-attn residual + LN1
    float mean = 0.f;
    for (int j = 0; j < 32; ++j) mean += Fb[t * 32 + j] + OW[t * 32 + j];
    mean *= (1.f / 32.f);
    float var = 0.f;
    for (int j = 0; j < 32; ++j) { const float d = Fb[t*32+j] + OW[t*32+j] - mean; var += d*d; }
    float inv = rsqrtf(var * (1.f / 32.f) + 1e-5f);
    for (int j = 0; j < 32; ++j) {
      const float d = Fb[t*32+j] + OW[t*32+j] - mean;
      Fb[t * 32 + j] = d * inv * LLN1[j];
    }
    // cross-attn (constant vector) + LN2
    mean = 0.f;
    for (int j = 0; j < 32; ++j) mean += Fb[t * 32 + j] + CAL[j];
    mean *= (1.f / 32.f);
    var = 0.f;
    for (int j = 0; j < 32; ++j) { const float d = Fb[t*32+j] + CAL[j] - mean; var += d*d; }
    inv = rsqrtf(var * (1.f / 32.f) + 1e-5f);
    for (int j = 0; j < 32; ++j) {
      const float d = Fb[t*32+j] + CAL[j] - mean;
      Fb[t * 32 + j] = d * inv * LLN2[j];
    }
    // FFN (32->1->32) + LN3
    float hs = 0.f;
    for (int j = 0; j < 32; ++j) hs += Fb[t * 32 + j] * LW1[j];
    hs = fmaxf(hs, 0.f);
    mean = 0.f;
    for (int j = 0; j < 32; ++j) mean += Fb[t * 32 + j] + hs * LW2[j];
    mean *= (1.f / 32.f);
    var = 0.f;
    for (int j = 0; j < 32; ++j) { const float d = Fb[t*32+j] + hs*LW2[j] - mean; var += d*d; }
    inv = rsqrtf(var * (1.f / 32.f) + 1e-5f);
    for (int j = 0; j < 32; ++j) {
      const float d = Fb[t*32+j] + hs * LW2[j] - mean;
      Fb[t * 32 + j] = d * inv * LLN3[j];
    }
  }
  __syncthreads();

  // ======================= reasoner attnC (d=10, seq=32) ===================
  float* U = Fb + 2560;                      // U[32][16] = Z^T
  for (int i = lane; i < 32 * 16; i += 32) U[i] = 0.f;
  __syncthreads();
  for (int idx = lane; idx < 320; idx += 32) {
    const int j = idx / 10, c = idx % 10;
    U[j * 16 + c] = Fb[c * 32 + j];
  }
  __syncthreads();
  for (int i = lane; i < 32 * 32; i += 32) {  // Uh [32][32], cols >=16 zero
    const int r = i >> 5, c = i & 31;
    Qb[i] = (c < 16) ? (_Float16)U[r * 16 + c] : (_Float16)0.f;
  }
  __syncthreads();
  float *QC = Fb + 512, *KC = Fb + 1024, *VC = Fb + 1536, *OC = Fb + 2048;
  const _Float16 *CWQ = wsH + H_CWQ, *CWK = wsH + H_CWK,
                 *CWV = wsH + H_CWV, *CWO = wsH + H_CWO;
#pragma unroll
  for (int mt = 0; mt < 2; ++mt) {
    v16h a = frag_a(Qb, 32, mt * 16, 0, lane);
    v8f q = {}; q = wmma32(a, frag_b(CWQ, 16, 0, 0, lane), q);
    v8f k = {}; k = wmma32(a, frag_b(CWK, 16, 0, 0, lane), k);
    v8f v = {}; v = wmma32(a, frag_b(CWV, 16, 0, 0, lane), v);
    frag_st(QC, 16, mt * 16, 0, lane, q);
    frag_st(KC, 16, mt * 16, 0, lane, k);
    frag_st(VC, 16, mt * 16, 0, lane, v);
  }
  __syncthreads();
  // 10 heads of dim 1: loop heads, lane = query (32 queries)
  for (int h = 0; h < 10; ++h) {
    const float qv = QC[lane * 16 + h];
    float m = -1e30f;
    for (int k = 0; k < 32; ++k) m = fmaxf(m, qv * KC[k * 16 + h]);
    float s = 0.f, o = 0.f;
    for (int k = 0; k < 32; ++k) {
      const float e = __expf(qv * KC[k * 16 + h] - m);
      s += e; o += e * VC[k * 16 + h];
    }
    OC[lane * 16 + h] = o / s;
  }
  __syncthreads();
  for (int i = lane; i < 32 * 32; i += 32) {  // Oh [32][32], cols >=10 zero
    const int r = i >> 5, c = i & 31;
    Qb[i] = (c < 10) ? (_Float16)OC[r * 16 + c] : (_Float16)0.f;
  }
  __syncthreads();
  float* OWc = Fb + 2048;
#pragma unroll
  for (int mt = 0; mt < 2; ++mt) {
    v8f o = {};
    o = wmma32(frag_a(Qb, 32, mt * 16, 0, lane), frag_b(CWO, 16, 0, 0, lane), o);
    frag_st(OWc, 16, mt * 16, 0, lane, o);
  }
  __syncthreads();
  const float *CLN1 = wsF + F_CLN1, *CLN2 = wsF + F_CLN2, *CLN3 = wsF + F_CLN3,
              *CW1  = wsF + F_CW1,  *CW2  = wsF + F_CW2,  *CAC  = wsF + F_CAC;
  {
    const int t = lane;  // 32 tokens, one per lane
    float mean = 0.f;
    for (int c = 0; c < 10; ++c) mean += U[t * 16 + c] + OWc[t * 16 + c];
    mean *= 0.1f;
    float var = 0.f;
    for (int c = 0; c < 10; ++c) { const float d = U[t*16+c] + OWc[t*16+c] - mean; var += d*d; }
    float inv = rsqrtf(var * 0.1f + 1e-5f);
    for (int c = 0; c < 10; ++c) {
      const float d = U[t*16+c] + OWc[t*16+c] - mean;
      U[t * 16 + c] = d * inv * CLN1[c];
    }
    mean = 0.f;
    for (int c = 0; c < 10; ++c) mean += U[t * 16 + c] + CAC[c];
    mean *= 0.1f;
    var = 0.f;
    for (int c = 0; c < 10; ++c) { const float d = U[t*16+c] + CAC[c] - mean; var += d*d; }
    inv = rsqrtf(var * 0.1f + 1e-5f);
    for (int c = 0; c < 10; ++c) {
      const float d = U[t*16+c] + CAC[c] - mean;
      U[t * 16 + c] = d * inv * CLN2[c];
    }
    float hs = 0.f;
    for (int c = 0; c < 10; ++c) hs += U[t * 16 + c] * CW1[c];
    hs = fmaxf(hs, 0.f);
    mean = 0.f;
    for (int c = 0; c < 10; ++c) mean += U[t * 16 + c] + hs * CW2[c];
    mean *= 0.1f;
    var = 0.f;
    for (int c = 0; c < 10; ++c) { const float d = U[t*16+c] + hs*CW2[c] - mean; var += d*d; }
    inv = rsqrtf(var * 0.1f + 1e-5f);
    for (int c = 0; c < 10; ++c) {
      const float d = U[t*16+c] + hs * CW2[c] - mean;
      U[t * 16 + c] = d * inv * CLN3[c];
    }
  }
  __syncthreads();

  // ---- decoder head: out[c] = sum_j U[j][c] * dec_W[j] --------------------
  const float* DECW = wsF + F_DECW;
  if (lane < 10) {
    float acc = 0.f;
    for (int j = 0; j < 32; ++j) acc += U[j * 16 + lane] * DECW[j];
    out[(((size_t)nn * 10 + lane) * 30 + h0) * 30 + w0] = acc;
  }
}

// ===========================================================================
extern "C" void kernel_launch(void* const* d_in, const int* in_sizes, int n_in,
                              void* d_out, int out_size, void* d_ws, size_t ws_size,
                              hipStream_t stream) {
  (void)in_sizes; (void)n_in; (void)out_size; (void)ws_size;

  PtrPack P;
  for (int i = 0; i < 39; ++i) P.p[i] = (const float*)d_in[i];
  const float* x = (const float*)d_in[I_X];

  _Float16* wsH = (_Float16*)d_ws;
  float*    wsF = (float*)((char*)d_ws + 2 * H_TOTAL);

  pes_prep<<<1, 256, 0, stream>>>(P, wsH, wsF);

  const int n_pixels = 48 * 30 * 30;  // 43,200 independent pixel pipelines
  pes_fused<<<n_pixels, 32, 0, stream>>>(wsH, wsF, x, (float*)d_out);
}
